// Attention_23862838296829
// MI455X (gfx1250) — matrix operand order repaired and checked by reading, not compiled, not run on hardware
//
#include <hip/hip_runtime.h>
#include <hip/hip_bf16.h>

typedef __bf16 bf16_t;
typedef __attribute__((ext_vector_type(8)))  __bf16 v8bf;
typedef __attribute__((ext_vector_type(16))) __bf16 v16bf;
typedef __attribute__((ext_vector_type(8)))  float  v8f;
typedef __attribute__((ext_vector_type(2)))  float  v2f;

#define T_SEQ 1024
#define C_DIM 768
#define N_BR  12
#define DH    64
#define H_TOT 144
#define CQ    9216   // C_DIM * N_BR

// ---------------------------------------------------------------------------
// WMMA helpers
// ---------------------------------------------------------------------------
__device__ __forceinline__ v8f wmma_bf16(v16bf a, v16bf b, v8f c) {
    return __builtin_amdgcn_wmma_f32_16x16x32_bf16(
        /*neg_a=*/false, a, /*neg_b=*/false, b,
        /*c_mod=*/(short)0, c, /*reuse_a=*/false, /*reuse_b=*/false);
}

__device__ __forceinline__ v8f wmma_f32(v2f a, v2f b, v8f c) {
    return __builtin_amdgcn_wmma_f32_16x16x4_f32(
        /*neg_a=*/false, a, /*neg_b=*/false, b,
        /*c_mod=*/(short)0, c, /*reuse_a=*/false, /*reuse_b=*/false);
}

// A-matrix fragment (16x32 bf16, MxK). Lane half l>>4 selects K block:
// lanes 0-15 : K = 0..7 and 16..23 ; lanes 16-31 : K = 8..15 and 24..31.
__device__ __forceinline__ v16bf load_fragA(const bf16_t* base, int rowStride, int kOff) {
    const int lane = __lane_id();
    const int m    = lane & 15;
    const int kb   = ((lane >> 4) << 3) + kOff;
    const bf16_t* p = base + (size_t)m * rowStride + kb;
    v8bf lo = *reinterpret_cast<const v8bf*>(p);
    v8bf hi = *reinterpret_cast<const v8bf*>(p + 16);
    return __builtin_shufflevector(lo, hi, 0,1,2,3,4,5,6,7,8,9,10,11,12,13,14,15);
}

// B-matrix fragment (32x16 bf16, KxN), staged transposed (N rows, K contiguous).
// lanes 0-15 : K = 0..7 and 16..23 ; lanes 16-31 : K = 8..15 and 24..31.
__device__ __forceinline__ v16bf load_fragB(const bf16_t* baseT, int rowStride, int kOff) {
    const int lane = __lane_id();
    const int n    = lane & 15;
    const int kb   = ((lane >> 4) << 3) + kOff;
    const bf16_t* p = baseT + (size_t)n * rowStride + kb;
    v8bf lo = *reinterpret_cast<const v8bf*>(p);
    v8bf hi = *reinterpret_cast<const v8bf*>(p + 16);
    return __builtin_shufflevector(lo, hi, 0,1,2,3,4,5,6,7,8,9,10,11,12,13,14,15);
}

// ---------------------------------------------------------------------------
// CDNA5 async global -> LDS staging (ASYNCcnt path).
// GV mode: per-lane 64-bit global address, per-lane 32-bit LDS byte offset.
// Generic pointers to __shared__ carry the LDS offset in their low 32 bits.
// ---------------------------------------------------------------------------
__device__ __forceinline__ void async_copy_b128(void* lds_ptr, const void* gptr) {
    unsigned lds_off = (unsigned)(uintptr_t)lds_ptr;
    asm volatile("global_load_async_to_lds_b128 %0, %1, off"
                 :: "v"(lds_off), "v"(gptr) : "memory");
}

__device__ __forceinline__ void wait_async() {
    asm volatile("s_wait_asynccnt 0x0" ::: "memory");
}

// ---------------------------------------------------------------------------
// f32 -> bf16 conversion (grid-stride)
// ---------------------------------------------------------------------------
__global__ void cvt_bf16_kernel(const float* __restrict__ src,
                                bf16_t* __restrict__ dst, int n) {
    int i = blockIdx.x * blockDim.x + threadIdx.x;
    int stride = gridDim.x * blockDim.x;
    for (; i < n; i += stride) dst[i] = (bf16_t)src[i];
}

// V reorder: (t, sh*64+d) f32  ->  (sh, t, d) bf16
__global__ void v_reorder_kernel(const float* __restrict__ Vf,
                                 bf16_t* __restrict__ Vb) {
    int i = blockIdx.x * blockDim.x + threadIdx.x;
    if (i >= N_BR * T_SEQ * DH) return;
    int d = i & 63;
    int t = (i >> 6) & (T_SEQ - 1);
    int s = i >> 16;                     // / (1024*64)
    Vb[i] = (bf16_t)Vf[(size_t)t * C_DIM + s * DH + d];
}

// mean over branches of WO_b
__global__ void mean_bias_kernel(const float* __restrict__ WO_b,
                                 float* __restrict__ mb) {
    int c = blockIdx.x * blockDim.x + threadIdx.x;
    if (c >= C_DIM) return;
    float s = 0.f;
    for (int n = 0; n < N_BR; ++n) s += WO_b[n * C_DIM + c];
    mb[c] = s * (1.0f / 12.0f);
}

// ---------------------------------------------------------------------------
// Tiled bf16 GEMM with f32 accumulate:  C[M,N] = scale * (A[M,K] @ B[K,N]) + bias
// Block tile 128x64, 8 waves in a 4x2 grid, each wave 32x32 (4 WMMA accs).
// A tile staged with async global->LDS; B tile staged transposed via VGPRs.
// ---------------------------------------------------------------------------
#define GTM 128
#define GTN 64
#define GTK 32

__global__ __launch_bounds__(256) void gemm_bf16_kernel(
    const bf16_t* __restrict__ A, const bf16_t* __restrict__ B,
    const float* __restrict__ bias, float* __restrict__ C,
    int M, int N, int K, float outScale)
{
    __shared__ bf16_t sA [GTM * GTK];   // 8 KB
    __shared__ bf16_t sBT[GTN * GTK];   // 4 KB, transposed (N rows, K contiguous)

    const int tid  = threadIdx.x;
    const int lane = tid & 31;
    const int wave = tid >> 5;
    const int wm   = wave & 3;
    const int wn   = wave >> 2;
    const int m0   = blockIdx.y * GTM;
    const int n0   = blockIdx.x * GTN;

    v8f acc[2][2];
#pragma unroll
    for (int i = 0; i < 2; ++i)
#pragma unroll
        for (int j = 0; j < 2; ++j)
#pragma unroll
            for (int r = 0; r < 8; ++r) acc[i][j][r] = 0.0f;

    const int aIdx = tid * 16;                // element index in 128x32 tile
    const int aRow = aIdx >> 5;
    const int aCol = aIdx & 31;

    for (int k0 = 0; k0 < K; k0 += GTK) {
        // stage A tile: async global -> LDS, 2 x b128 per thread
        {
            const bf16_t* g = A + (size_t)(m0 + aRow) * K + k0 + aCol;
            async_copy_b128(&sA[aIdx],     g);
            async_copy_b128(&sA[aIdx + 8], g + 8);
            if (k0 + GTK < K) __builtin_prefetch(g + GTK, 0, 1);
        }
        // stage B^T tile: read B (KxN row-major) coalesced, write transposed
        {
            int c = tid & 63;           // n within tile
            int rb = tid >> 6;          // 0..3
#pragma unroll
            for (int i = 0; i < 8; ++i) {
                int r = rb + i * 4;     // k within tile 0..31
                sBT[c * GTK + r] = B[(size_t)(k0 + r) * N + (n0 + c)];
            }
        }
        wait_async();
        __syncthreads();

        v16bf a0 = load_fragA(sA  + (wm * 32)      * GTK, GTK, 0);
        v16bf a1 = load_fragA(sA  + (wm * 32 + 16) * GTK, GTK, 0);
        v16bf b0 = load_fragB(sBT + (wn * 32)      * GTK, GTK, 0);
        v16bf b1 = load_fragB(sBT + (wn * 32 + 16) * GTK, GTK, 0);
        acc[0][0] = wmma_bf16(a0, b0, acc[0][0]);
        acc[0][1] = wmma_bf16(a0, b1, acc[0][1]);
        acc[1][0] = wmma_bf16(a1, b0, acc[1][0]);
        acc[1][1] = wmma_bf16(a1, b1, acc[1][1]);
        __syncthreads();
    }

#pragma unroll
    for (int mf = 0; mf < 2; ++mf)
#pragma unroll
        for (int nf = 0; nf < 2; ++nf) {
            int row = m0 + wm * 32 + mf * 16 + ((lane >> 4) << 3);
            int col = n0 + wn * 32 + nf * 16 + (lane & 15);
            float bv = bias ? bias[col] : 0.0f;
#pragma unroll
            for (int r = 0; r < 8; ++r)
                C[(size_t)(row + r) * N + col] = acc[mf][nf][r] * outScale + bv;
        }
}

// ---------------------------------------------------------------------------
// Wedge (x + x @ (A - A^T + diag(id_bias_h))) via V_WMMA_F32_16X16X4_F32,
// followed by RoPE; writes bf16 in head-major (h, t, dh) layout.
// One block = one head x 64 sequence rows; 8 waves, each owning two 16x16
// output tiles of the 64x64 result.
// ---------------------------------------------------------------------------
__global__ __launch_bounds__(256) void wedge_rope_kernel(
    const float* __restrict__ src, bf16_t* __restrict__ dst,
    const float* __restrict__ Amat, const float* __restrict__ id_bias,
    int srcStride, int isK, float outScale)
{
    __shared__ float sS[DH * DH];
    __shared__ float sX[DH * DH];
    __shared__ float sY[DH * DH];

    const int h   = blockIdx.y;
    const int t0  = blockIdx.x * 64;
    const int tid = threadIdx.x;
    const int lane = tid & 31;
    const int wave = tid >> 5;
    const int colBase = (isK ? (h % N_BR) : h) * DH;

#pragma unroll
    for (int i = 0; i < 16; ++i) {
        int idx = tid + i * 256;
        int d = idx >> 6, e = idx & 63;
        float s = Amat[d * DH + e] - Amat[e * DH + d];
        if (d == e) s += id_bias[h * DH + d];
        sS[idx] = s;
        sX[idx] = src[(size_t)(t0 + d) * srcStride + colBase + e];
    }
    __syncthreads();

    // flow = X @ S using f32 WMMA (16x16x4).
    // A frag: a[j] = X[m][kb + j], kb = k0 + (lane>>4)*2
    // B frag: b[j] = S[kb + j][n]
    const int mt = wave & 3;            // M tile index (rows mt*16..)
    const int eb = (wave >> 2) * 2;     // first of two E tiles
    const int mrow = mt * 16 + (lane & 15);
    const int ncol0 = eb * 16 + (lane & 15);
    const int ncol1 = ncol0 + 16;

    v8f fl[2];
#pragma unroll
    for (int j = 0; j < 2; ++j)
#pragma unroll
        for (int r = 0; r < 8; ++r) fl[j][r] = 0.0f;

#pragma unroll
    for (int k0 = 0; k0 < DH; k0 += 4) {
        int kb = k0 + ((lane >> 4) << 1);
        v2f a  = { sX[mrow * DH + kb],   sX[mrow * DH + kb + 1] };
        v2f b0 = { sS[kb * DH + ncol0],  sS[(kb + 1) * DH + ncol0] };
        v2f b1 = { sS[kb * DH + ncol1],  sS[(kb + 1) * DH + ncol1] };
        fl[0] = wmma_f32(a, b0, fl[0]);
        fl[1] = wmma_f32(a, b1, fl[1]);
    }

    // Y = X + flow, scattered back to LDS in C-layout order
#pragma unroll
    for (int j = 0; j < 2; ++j) {
        int e = (eb + j) * 16 + (lane & 15);
#pragma unroll
        for (int r = 0; r < 8; ++r) {
            int row = mt * 16 + ((lane >> 4) << 3) + r;
            sY[row * DH + e] = fl[j][r] + sX[row * DH + e];
        }
    }
    __syncthreads();

    // RoPE
#pragma unroll
    for (int i = 0; i < 8; ++i) {
        int idx = tid + i * 256;        // 0..2047 : (row, freq pair)
        int r = idx >> 5, j = idx & 31;
        float tpos = (float)(t0 + r);
        float inv  = __powf(10000.0f, -(float)(2 * j) * (1.0f / 64.0f));
        float f    = tpos * inv;
        float cs   = __cosf(f), sn = __sinf(f);
        float x1 = sY[r * DH + 2 * j];
        float x2 = sY[r * DH + 2 * j + 1];
        size_t ob = ((size_t)h * T_SEQ + (t0 + r)) * DH;
        dst[ob + j]      = (bf16_t)((x1 * cs - x2 * sn) * outScale);
        dst[ob + 32 + j] = (bf16_t)((x1 * sn + x2 * cs) * outScale);
    }
}

// ---------------------------------------------------------------------------
// Causal flash attention with sink column. One block = (head, 128 q rows);
// 8 waves, each wave owns a 16x64 output fragment. Key tiles of 64.
// K tile staged with async global->LDS.
// ---------------------------------------------------------------------------
__global__ __launch_bounds__(256) void flash_attn_kernel(
    const bf16_t* __restrict__ Qp, const bf16_t* __restrict__ Kp,
    const bf16_t* __restrict__ Vb, const float* __restrict__ sink,
    const float* __restrict__ v_nulls, bf16_t* __restrict__ ctx)
{
    __shared__ bf16_t sK [64 * 64];       // 8 KB : key tile (key, dh)
    __shared__ bf16_t sVT[64 * 64];       // 8 KB : V^T tile (dh, key)
    __shared__ bf16_t sP [8][16 * 64];    // 16 KB: per-wave probs (row, key)

    const int h     = blockIdx.y;
    const int q0    = blockIdx.x * 128;
    const int tid   = threadIdx.x;
    const int lane  = tid & 31;
    const int wave  = tid >> 5;
    const int qrow0 = q0 + wave * 16;
    const int sh    = h % N_BR;

    // Q fragments (global, K-contiguous rows) for dh-steps 0 and 32
    const bf16_t* qbase = Qp + ((size_t)h * T_SEQ + qrow0) * DH;
    v16bf qa0 = load_fragA(qbase, DH, 0);
    v16bf qa1 = load_fragA(qbase, DH, 32);

    float m_run[8], l_run[8];
    v8f acc[4];
    const float s0 = sink[h];
#pragma unroll
    for (int r = 0; r < 8; ++r) { m_run[r] = s0; l_run[r] = 1.0f; }
#pragma unroll
    for (int nf = 0; nf < 4; ++nf) {
        float vn = v_nulls[h * DH + nf * 16 + (lane & 15)];
#pragma unroll
        for (int r = 0; r < 8; ++r) acc[nf][r] = vn;   // sink contribution
    }

    const int nkt = blockIdx.x * 2 + 2;   // 64-key tiles covering [0, q0+128)
    for (int kt = 0; kt < nkt; ++kt) {
        __syncthreads();
        // stage K tile (rows contiguous in (h,t,dh) layout): async -> LDS
        {
            int idx = tid * 16;
            const bf16_t* g = Kp + ((size_t)h * T_SEQ + kt * 64) * DH + idx;
            async_copy_b128(&sK[idx],     g);
            async_copy_b128(&sK[idx + 8], g + 8);
            if (kt + 1 < nkt) __builtin_prefetch(g + 64 * DH, 0, 1);
        }
        // stage V^T tile
#pragma unroll
        for (int i = 0; i < 16; ++i) {
            int idx = tid + i * 256;
            int s = idx & 63, d = idx >> 6;
            sVT[d * 64 + s] =
                Vb[((size_t)sh * T_SEQ + kt * 64 + s) * DH + d];
        }
        wait_async();
        __syncthreads();

        // scores: 16 q-rows x 64 keys
        v8f sc[4];
#pragma unroll
        for (int nf = 0; nf < 4; ++nf) {
#pragma unroll
            for (int r = 0; r < 8; ++r) sc[nf][r] = 0.0f;
            v16bf b0 = load_fragB(sK + (nf * 16) * 64, 64, 0);
            v16bf b1 = load_fragB(sK + (nf * 16) * 64, 64, 32);
            sc[nf] = wmma_bf16(qa0, b0, sc[nf]);
            sc[nf] = wmma_bf16(qa1, b1, sc[nf]);
        }
        // causal mask
#pragma unroll
        for (int nf = 0; nf < 4; ++nf) {
            int keyn = kt * 64 + nf * 16 + (lane & 15);
#pragma unroll
            for (int r = 0; r < 8; ++r) {
                int qr = qrow0 + ((lane >> 4) << 3) + r;
                if (keyn > qr) sc[nf][r] = -1e30f;
            }
        }
        // online softmax (16 lanes share a row; reduce within lane-half)
#pragma unroll
        for (int r = 0; r < 8; ++r) {
            float mx = sc[0][r];
#pragma unroll
            for (int nf = 1; nf < 4; ++nf) mx = fmaxf(mx, sc[nf][r]);
            for (int off = 1; off < 16; off <<= 1)
                mx = fmaxf(mx, __shfl_xor(mx, off, 32));
            float mnew = fmaxf(m_run[r], mx);
            float corr = __expf(m_run[r] - mnew);
            float psum = 0.0f;
#pragma unroll
            for (int nf = 0; nf < 4; ++nf) {
                float p = __expf(sc[nf][r] - mnew);
                sc[nf][r] = p;
                psum += p;
            }
            for (int off = 1; off < 16; off <<= 1)
                psum += __shfl_xor(psum, off, 32);
            l_run[r] = l_run[r] * corr + psum;
            m_run[r] = mnew;
#pragma unroll
            for (int nf = 0; nf < 4; ++nf) acc[nf][r] *= corr;
        }
        // P: C-layout -> A-layout via per-wave LDS staging
        bf16_t* pw = sP[wave];
#pragma unroll
        for (int nf = 0; nf < 4; ++nf) {
            int kl = nf * 16 + (lane & 15);
#pragma unroll
            for (int r = 0; r < 8; ++r) {
                int m = ((lane >> 4) << 3) + r;
                pw[m * 64 + kl] = (bf16_t)sc[nf][r];
            }
        }
        __syncthreads();
        // P @ V
        v16bf pa0 = load_fragA(pw, 64, 0);
        v16bf pa1 = load_fragA(pw, 64, 32);
#pragma unroll
        for (int nf = 0; nf < 4; ++nf) {
            v16bf bv0 = load_fragB(sVT + (nf * 16) * 64, 64, 0);
            v16bf bv1 = load_fragB(sVT + (nf * 16) * 64, 64, 32);
            acc[nf] = wmma_bf16(pa0, bv0, acc[nf]);
            acc[nf] = wmma_bf16(pa1, bv1, acc[nf]);
        }
    }

    // normalize + scatter into ctx (t, n*768 + sh*64 + d) bf16
    const int nbr = h / N_BR;
#pragma unroll
    for (int nf = 0; nf < 4; ++nf) {
#pragma unroll
        for (int r = 0; r < 8; ++r) {
            int t   = qrow0 + ((lane >> 4) << 3) + r;
            int col = nbr * C_DIM + sh * DH + nf * 16 + (lane & 15);
            ctx[(size_t)t * CQ + col] = (bf16_t)(acc[nf][r] / l_run[r]);
        }
    }
}

// ---------------------------------------------------------------------------
// Host-side launch
// ---------------------------------------------------------------------------
extern "C" void kernel_launch(void* const* d_in, const int* in_sizes, int n_in,
                              void* d_out, int out_size, void* d_ws, size_t ws_size,
                              hipStream_t stream) {
    (void)in_sizes; (void)n_in; (void)out_size; (void)ws_size;

    const float* x       = (const float*)d_in[0];
    const float* WV_w    = (const float*)d_in[1];
    const float* WV_b    = (const float*)d_in[2];
    const float* WK_w    = (const float*)d_in[3];
    const float* WK_b    = (const float*)d_in[4];
    const float* WQ_w    = (const float*)d_in[5];
    const float* Amat    = (const float*)d_in[6];
    const float* id_bias = (const float*)d_in[7];
    const float* sink    = (const float*)d_in[8];
    const float* v_nulls = (const float*)d_in[9];
    const float* WO_w    = (const float*)d_in[10];
    const float* WO_b    = (const float*)d_in[11];
    float* out = (float*)d_out;

    char* ws = (char*)d_ws;
    size_t off = 0;
    auto alloc = [&](size_t bytes) -> void* {
        void* p = (void*)(ws + off);
        off = (off + bytes + 255) & ~(size_t)255;
        return p;
    };

    bf16_t* xb  = (bf16_t*)alloc((size_t)T_SEQ * C_DIM * 2);
    bf16_t* WQb = (bf16_t*)alloc((size_t)C_DIM * CQ * 2);
    bf16_t* WKb = (bf16_t*)alloc((size_t)C_DIM * C_DIM * 2);
    bf16_t* WVb = (bf16_t*)alloc((size_t)C_DIM * C_DIM * 2);
    bf16_t* WOb = (bf16_t*)alloc((size_t)N_BR * C_DIM * C_DIM * 2);
    float*  Qf  = (float*)alloc((size_t)T_SEQ * CQ * 4);
    float*  Kf  = (float*)alloc((size_t)T_SEQ * C_DIM * 4);
    float*  Vf  = (float*)alloc((size_t)T_SEQ * C_DIM * 4);
    bf16_t* Qp  = (bf16_t*)alloc((size_t)H_TOT * T_SEQ * DH * 2);
    bf16_t* Kp  = (bf16_t*)alloc((size_t)H_TOT * T_SEQ * DH * 2);
    bf16_t* Vbf = (bf16_t*)alloc((size_t)N_BR * T_SEQ * DH * 2);
    bf16_t* ctx = (bf16_t*)alloc((size_t)T_SEQ * CQ * 2);
    float*  mb  = (float*)alloc((size_t)C_DIM * 4);

    auto cvt = [&](const float* s, bf16_t* d, int n) {
        int blocks = (n + 1023) / 1024;
        cvt_bf16_kernel<<<blocks, 256, 0, stream>>>(s, d, n);
    };
    cvt(x,    xb,  T_SEQ * C_DIM);
    cvt(WQ_w, WQb, C_DIM * CQ);
    cvt(WK_w, WKb, C_DIM * C_DIM);
    cvt(WV_w, WVb, C_DIM * C_DIM);
    cvt(WO_w, WOb, N_BR * C_DIM * C_DIM);

    // Projections
    gemm_bf16_kernel<<<dim3(CQ / GTN, T_SEQ / GTM), 256, 0, stream>>>(
        xb, WQb, nullptr, Qf, T_SEQ, CQ, C_DIM, 1.0f);
    gemm_bf16_kernel<<<dim3(C_DIM / GTN, T_SEQ / GTM), 256, 0, stream>>>(
        xb, WKb, WK_b, Kf, T_SEQ, C_DIM, C_DIM, 1.0f);
    gemm_bf16_kernel<<<dim3(C_DIM / GTN, T_SEQ / GTM), 256, 0, stream>>>(
        xb, WVb, WV_b, Vf, T_SEQ, C_DIM, C_DIM, 1.0f);

    v_reorder_kernel<<<(N_BR * T_SEQ * DH) / 256, 256, 0, stream>>>(Vf, Vbf);

    // Wedge + RoPE (q pre-scaled by dh^-0.5 = 0.125)
    wedge_rope_kernel<<<dim3(T_SEQ / 64, H_TOT), 256, 0, stream>>>(
        Qf, Qp, Amat, id_bias, CQ, 0, 0.125f);
    wedge_rope_kernel<<<dim3(T_SEQ / 64, H_TOT), 256, 0, stream>>>(
        Kf, Kp, Amat, id_bias, C_DIM, 1, 1.0f);

    // Attention
    flash_attn_kernel<<<dim3(T_SEQ / 128, H_TOT), 256, 0, stream>>>(
        Qp, Kp, Vbf, sink, v_nulls, ctx);

    // Output projection: out = (1/12) * ctx(1024x9216) @ WO(9216x768) + mean(WO_b)
    mean_bias_kernel<<<3, 256, 0, stream>>>(WO_b, mb);
    gemm_bf16_kernel<<<dim3(C_DIM / GTN, T_SEQ / GTM), 256, 0, stream>>>(
        ctx, WOb, mb, out, T_SEQ, C_DIM, CQ, 1.0f / 12.0f);
}
